// S2V_modified_33260226740722
// MI455X (gfx1250) — matrix operand order repaired and checked by reading, not compiled, not run on hardware
//
#include <hip/hip_runtime.h>
#include <hip/hip_bf16.h>

typedef __attribute__((ext_vector_type(16))) _Float16 v16h;
typedef __attribute__((ext_vector_type(8)))  _Float16 v8h;
typedef __attribute__((ext_vector_type(4)))  _Float16 v4h;
typedef __attribute__((ext_vector_type(8)))  float    v8f;

#define B_ 32
#define N_ 1024
#define F_ 32
#define D_ 64
#define H_ 64

__device__ __forceinline__ v8f wmma_f16(v16h a, v16h b, v8f c) {
  // D = A(16x32 f16) * B(32x16 f16) + C(16x16 f32)
  return __builtin_amdgcn_wmma_f32_16x16x32_f16(
      /*neg_a=*/false, a, /*neg_b=*/false, b,
      /*c_mod=*/(short)0, c, /*reuse_a=*/false, /*reuse_b=*/false);
}

// ---------------------------------------------------------------------------
// adj (f32) -> binary f16 adjacency (exact: values are 0.0h / 1.0h)
// ---------------------------------------------------------------------------
__global__ void k_prep(const float* __restrict__ adj, _Float16* __restrict__ Ah) {
  size_t i = (size_t)blockIdx.x * 256 + threadIdx.x;
  if (i < (size_t)N_ * N_)
    Ah[i] = (adj[i] > 0.f) ? (_Float16)1.0f : (_Float16)0.0f;
}

// ---------------------------------------------------------------------------
// mu_1 = relu(xv @ mu1) : [B*N, F=32] @ [32, D=64]   (0.13 GF, VALU)
// ---------------------------------------------------------------------------
__global__ void k_mu1(const float* __restrict__ xv, const float* __restrict__ w,
                      float* __restrict__ out) {
  size_t i = (size_t)blockIdx.x * 256 + threadIdx.x;
  if (i >= (size_t)B_ * N_ * D_) return;
  int d = (int)(i & (D_ - 1));
  size_t row = i >> 6;                       // b*N + n
  const float* x = xv + row * F_;
  float acc = 0.f;
#pragma unroll
  for (int f = 0; f < F_; ++f) acc += x[f] * w[f * D_ + d];
  out[i] = fmaxf(acc, 0.f);
}

// ---------------------------------------------------------------------------
// mu4[n,d] = sum_m relu(adj[n,m]*W4[d] + b4[d]);  mu3 = mu4 @ W3^T + b3
// ---------------------------------------------------------------------------
__global__ void k_mu4mu3(const float* __restrict__ adj, const float* __restrict__ W4,
                         const float* __restrict__ b4, const float* __restrict__ W3,
                         const float* __restrict__ b3, float* __restrict__ mu3) {
  int n = blockIdx.x, d = threadIdx.x;
  float w4 = W4[d], bb = b4[d];
  const float* arow = adj + (size_t)n * N_;
  float acc = 0.f;
  for (int m = 0; m < N_; ++m) acc += fmaxf(arow[m] * w4 + bb, 0.f);
  __shared__ float mu4s[D_];
  mu4s[d] = acc;
  __syncthreads();
  float a2 = b3[d];
#pragma unroll
  for (int dp = 0; dp < D_; ++dp) a2 += mu4s[dp] * W3[d * D_ + dp];
  mu3[(size_t)n * D_ + d] = a2;
}

// ---------------------------------------------------------------------------
// P_b = A @ mu_b  (dominant 4.3 GF GEMM over all batches) via WMMA f16->f32.
// grid = (N/64, B), block = 128 (4 wave32). 64x64 output per block.
// K streamed in 64-tiles, double-buffered LDS with register staging.
// mu tile stored TRANSPOSED in LDS so B fragments are 2x ds_load_b128.
// ---------------------------------------------------------------------------
__device__ __forceinline__ void spmm_load_regs(const _Float16* __restrict__ Ah,
                                               const float* __restrict__ muB,
                                               int n0, int k0, int tid,
                                               uint4 ra[4], float4 rb[8]) {
#pragma unroll
  for (int j = 0; j < 4; ++j) {            // A tile: 64 rows x 64 halves
    int idx = j * 128 + tid;               // 0..511 uint4 units
    int r = idx >> 3, cp = idx & 7;
    ra[j] = ((const uint4*)(Ah + (size_t)(n0 + r) * N_ + k0))[cp];
  }
#pragma unroll
  for (int j = 0; j < 8; ++j) {            // mu tile: 64 k x 64 d floats
    int idx = j * 128 + tid;               // 0..1023 float4 units
    int k = idx >> 4, c4 = idx & 15;
    rb[j] = ((const float4*)(muB + (size_t)(k0 + k) * D_))[c4];
  }
}

__device__ __forceinline__ void spmm_store_lds(_Float16 (*lA)[72], _Float16 (*lBT)[72],
                                               int tid, const uint4 ra[4],
                                               const float4 rb[8]) {
#pragma unroll
  for (int j = 0; j < 4; ++j) {
    int idx = j * 128 + tid;
    int r = idx >> 3, cp = idx & 7;
    *(uint4*)&lA[r][cp * 8] = ra[j];       // 16B-aligned (pitch 144B)
  }
#pragma unroll
  for (int j = 0; j < 8; ++j) {            // transposed commit: lBT[d][k]
    int idx = j * 128 + tid;
    int k = idx >> 4, c4 = (idx & 15) * 4;
    lBT[c4 + 0][k] = (_Float16)rb[j].x;
    lBT[c4 + 1][k] = (_Float16)rb[j].y;
    lBT[c4 + 2][k] = (_Float16)rb[j].z;
    lBT[c4 + 3][k] = (_Float16)rb[j].w;
  }
}

__device__ __forceinline__ void spmm_compute(const _Float16 (*lA)[72],
                                             const _Float16 (*lBT)[72],
                                             int mrow, int kh, int bcol, v8f acc[4]) {
#pragma unroll
  for (int kc = 0; kc < 64; kc += 32) {
    // A operand (16x32): lane<16 -> K0-15, lane>=16 -> K16-31
    union { v16h v; v8h p[2]; } af;
    af.p[0] = *(const v8h*)&lA[mrow][kc + kh * 16];
    af.p[1] = *(const v8h*)&lA[mrow][kc + kh * 16 + 8];
#pragma unroll
    for (int t = 0; t < 4; ++t) {
      // B operand (32x16): col = lane&15; lane<16 -> K0-15, lane>=16 -> K16-31
      union { v16h v; v8h p[2]; } bf;
      bf.p[0] = *(const v8h*)&lBT[t * 16 + bcol][kc + kh * 16];
      bf.p[1] = *(const v8h*)&lBT[t * 16 + bcol][kc + kh * 16 + 8];
      acc[t] = wmma_f16(af.v, bf.v, acc[t]);
    }
  }
}

__global__ void k_spmm(const _Float16* __restrict__ Ah,
                       const float* __restrict__ mu,
                       float* __restrict__ P) {
  __shared__ _Float16 lA[2][64][72];       // A rows (row-major), 144B pitch
  __shared__ _Float16 lBT[2][64][72];      // mu tile transposed: [d][k]
  const int b    = blockIdx.y;
  const int n0   = blockIdx.x * 64;
  const int tid  = threadIdx.x;
  const int wave = tid >> 5;
  const int lane = tid & 31;
  const int kh   = lane >> 4;
  const int bcol = lane & 15;
  const int mrow = (wave << 4) + bcol;
  const float* muB = mu + (size_t)b * (N_ * D_);

  v8f acc[4] = {};
  uint4 ra[4]; float4 rb[8];

  spmm_load_regs(Ah, muB, n0, 0, tid, ra, rb);
  spmm_store_lds(lA[0], lBT[0], tid, ra, rb);
  __syncthreads();

  const int NT = N_ / 64;
  for (int kt = 0; kt < NT; ++kt) {
    const int cur = kt & 1;
    if (kt + 1 < NT) spmm_load_regs(Ah, muB, n0, (kt + 1) * 64, tid, ra, rb);
    spmm_compute(lA[cur], lBT[cur], mrow, kh, bcol, acc);
    if (kt + 1 < NT) spmm_store_lds(lA[cur ^ 1], lBT[cur ^ 1], tid, ra, rb);
    __syncthreads();
  }

  // C/D layout: VGPR j, lane L -> row = j + 8*(L>=16), col = L&15
  const int rbase = n0 + (wave << 4) + (kh << 3);
#pragma unroll
  for (int t = 0; t < 4; ++t) {
    union { v8f v; float f[8]; } u; u.v = acc[t];
#pragma unroll
    for (int j = 0; j < 8; ++j)
      P[((size_t)b * N_ + rbase + j) * D_ + t * 16 + bcol] = u.f[j];
  }
}

// ---------------------------------------------------------------------------
// mu_out = relu(mu_1 + P @ W2^T + b2 + mu3)  via WMMA (K = 64, two chunks)
// grid = B*N/64 = 512, block = 128. Never reads previous mu => in-place safe.
// ---------------------------------------------------------------------------
__global__ void k_affine(const float* __restrict__ P,
                         const float* __restrict__ mu_1,
                         const float* __restrict__ mu3,
                         const float* __restrict__ W2,
                         const float* __restrict__ b2,
                         float* __restrict__ mu_out) {
  __shared__ _Float16 lP[64][72];          // 64 rows x 64 K(d')
  __shared__ _Float16 lW[64][72];          // lW[j][k] = W2[j,k]; Bop[k][j]=W2[j][k]
  const int r0   = blockIdx.x * 64;        // global row in [0, B*N)
  const int tid  = threadIdx.x;
  const int wave = tid >> 5;
  const int lane = tid & 31;
  const int kh   = lane >> 4;
  const int bcol = lane & 15;
  const int mrow = (wave << 4) + bcol;

#pragma unroll
  for (int j = 0; j < 8; ++j) {
    int idx = j * 128 + tid;               // 0..1023 float4 units
    int r = idx >> 4, c4 = (idx & 15) * 4;
    float4 p  = ((const float4*)(P  + (size_t)(r0 + r) * D_))[c4 >> 2];
    float4 w2 = ((const float4*)(W2 + (size_t)r * D_))[c4 >> 2];
    v4h hp, hw;
    hp[0] = (_Float16)p.x;  hp[1] = (_Float16)p.y;
    hp[2] = (_Float16)p.z;  hp[3] = (_Float16)p.w;
    hw[0] = (_Float16)w2.x; hw[1] = (_Float16)w2.y;
    hw[2] = (_Float16)w2.z; hw[3] = (_Float16)w2.w;
    *(v4h*)&lP[r][c4] = hp;                // 8B-aligned
    *(v4h*)&lW[r][c4] = hw;
  }
  __syncthreads();

  v8f acc[4] = {};
#pragma unroll
  for (int k0 = 0; k0 < D_; k0 += 32) {
    union { v16h v; v8h p[2]; } af;
    af.p[0] = *(const v8h*)&lP[mrow][k0 + kh * 16];
    af.p[1] = *(const v8h*)&lP[mrow][k0 + kh * 16 + 8];
#pragma unroll
    for (int t = 0; t < 4; ++t) {
      union { v16h v; v8h p[2]; } bf;
      bf.p[0] = *(const v8h*)&lW[t * 16 + bcol][k0 + kh * 16];
      bf.p[1] = *(const v8h*)&lW[t * 16 + bcol][k0 + kh * 16 + 8];
      acc[t] = wmma_f16(af.v, bf.v, acc[t]);
    }
  }

  const int rloc = (wave << 4) + (kh << 3);
#pragma unroll
  for (int t = 0; t < 4; ++t) {
    union { v8f v; float f[8]; } u; u.v = acc[t];
    const int d = t * 16 + bcol;
    const float bias = b2[d];
#pragma unroll
    for (int j = 0; j < 8; ++j) {
      int rg = r0 + rloc + j;
      int n  = rg & (N_ - 1);              // 64 | N => block stays in one batch
      float v = mu_1[(size_t)rg * D_ + d] + u.f[j] + bias + mu3[(size_t)n * D_ + d];
      mu_out[(size_t)rg * D_ + d] = fmaxf(v, 0.f);
    }
  }
}

// ---------------------------------------------------------------------------
// mu_mean[b] = relu(mean_n(mu[b]) @ Wq1^T + bq1) : grid B, block 64 (d)
// ---------------------------------------------------------------------------
__global__ void k_mean(const float* __restrict__ mu, const float* __restrict__ Wq1,
                       const float* __restrict__ bq1, float* __restrict__ mu_mean) {
  int b = blockIdx.x, d = threadIdx.x;
  const float* m = mu + (size_t)b * N_ * D_;
  float acc = 0.f;
  for (int n = 0; n < N_; ++n) acc += m[(size_t)n * D_ + d];
  __shared__ float s[D_];
  s[d] = acc * (1.0f / N_);
  __syncthreads();
  float a2 = bq1[d];
#pragma unroll
  for (int dp = 0; dp < D_; ++dp) a2 += s[dp] * Wq1[d * D_ + dp];
  mu_mean[b * D_ + d] = fmaxf(a2, 0.f);
}

// ---------------------------------------------------------------------------
// head: qreg = relu([row | q2] @ Wreg^T + breg); out = qreg @ Wq^T + bq
// grid = (N+1, B), block 64 (one thread per hidden k)
// ---------------------------------------------------------------------------
__global__ void k_head(const float* __restrict__ mu, const float* __restrict__ mu_mean,
                       const float* __restrict__ option, const float* __restrict__ Wq2,
                       const float* __restrict__ bq2, const float* __restrict__ Wreg,
                       const float* __restrict__ breg, const float* __restrict__ Wq,
                       const float* __restrict__ bq, float* __restrict__ out) {
  int b = blockIdx.y, row = blockIdx.x, k = threadIdx.x;
  const float* mrow = (row < N_) ? (mu + ((size_t)b * N_ + row) * D_)
                                 : (mu_mean + (size_t)b * D_);
  float opt = option[b];
  const float* wr = Wreg + (size_t)k * (2 * D_);
  float acc = breg[k];
#pragma unroll 8
  for (int d = 0; d < D_; ++d) acc += mrow[d] * wr[d];
#pragma unroll 8
  for (int d = 0; d < D_; ++d) acc += (opt * Wq2[d] + bq2[d]) * wr[D_ + d];
  __shared__ float red[H_];
  red[k] = fmaxf(acc, 0.f) * Wq[k];
  __syncthreads();
  if (k == 0) {
    float s = bq[0];
#pragma unroll
    for (int i = 0; i < H_; ++i) s += red[i];
    out[(size_t)b * (N_ + 1) + row] = s;
  }
}

// ---------------------------------------------------------------------------
extern "C" void kernel_launch(void* const* d_in, const int* in_sizes, int n_in,
                              void* d_out, int out_size, void* d_ws, size_t ws_size,
                              hipStream_t stream) {
  const float* xv     = (const float*)d_in[0];
  const float* option = (const float*)d_in[1];
  const float* adj    = (const float*)d_in[2];
  const float* mu1w   = (const float*)d_in[3];
  const float* W2     = (const float*)d_in[4];
  const float* b2     = (const float*)d_in[5];
  const float* W3     = (const float*)d_in[6];
  const float* b3     = (const float*)d_in[7];
  const float* W4     = (const float*)d_in[8];
  const float* b4     = (const float*)d_in[9];
  const float* Wq1    = (const float*)d_in[10];
  const float* bq1    = (const float*)d_in[11];
  const float* Wq2    = (const float*)d_in[12];
  const float* bq2    = (const float*)d_in[13];
  const float* Wreg   = (const float*)d_in[14];
  const float* breg   = (const float*)d_in[15];
  const float* Wq     = (const float*)d_in[16];
  const float* bq     = (const float*)d_in[17];
  float* out = (float*)d_out;

  // workspace carve-up (all L2-resident; ~28 MB total)
  char* w = (char*)d_ws;
  _Float16* Ah    = (_Float16*)(w);                     // 2 MiB  (N*N f16)
  float* mu_1     = (float*)(w + (2u  << 20));          // 8 MiB  (B*N*D f32)
  float* mu_buf   = (float*)(w + (10u << 20));          // 8 MiB
  float* P        = (float*)(w + (18u << 20));          // 8 MiB
  float* mu3      = (float*)(w + (26u << 20));          // 256 KiB (N*D f32)
  float* mu_mean  = (float*)(w + (27u << 20));          // 8 KiB   (B*D f32)

  k_prep<<<(N_ * N_ + 255) / 256, 256, 0, stream>>>(adj, Ah);
  k_mu1<<<(B_ * N_ * D_ + 255) / 256, 256, 0, stream>>>(xv, mu1w, mu_1);
  k_mu4mu3<<<N_, D_, 0, stream>>>(adj, W4, b4, W3, b3, mu3);

  for (int it = 0; it < 3; ++it) {
    const float* src = (it == 0) ? mu_1 : mu_buf;
    k_spmm<<<dim3(N_ / 64, B_), 128, 0, stream>>>(Ah, src, P);
    k_affine<<<(B_ * N_) / 64, 128, 0, stream>>>(P, mu_1, mu3, W2, b2, mu_buf);
  }

  k_mean<<<B_, D_, 0, stream>>>(mu_buf, Wq1, bq1, mu_mean);
  k_head<<<dim3(N_ + 1, B_), H_, 0, stream>>>(mu_buf, mu_mean, option, Wq2, bq2,
                                              Wreg, breg, Wq, bq, out);
}
